// Qwen3_5GatedDeltaNet_6734508720604
// MI455X (gfx1250) — matrix operand order repaired and checked by reading, not compiled, not run on hardware
//
#include <hip/hip_runtime.h>
#include <cstdint>

// Problem constants (fixed by the reference harness)
#define BB 2
#define SS 4096
#define HH 16
#define DK 128
#define DV 128
#define CC 64        // chunk size
#define NC 64        // number of chunks = SS/CC
#define NBH (BB*HH)  // 32 independent (b,h) recurrences

typedef _Float16 half_t;
typedef __attribute__((ext_vector_type(8)))  _Float16 v8h;
typedef __attribute__((ext_vector_type(16))) _Float16 v16h;
typedef __attribute__((ext_vector_type(8)))  float    v8f;

// -------------------------------------------------------------------------
// WMMA helpers (CDNA5 wave32, V_WMMA_F32_16X16X32_F16)
// D(16x16,f32) = A(16x32,f16) x B(32x16,f16) + C
// A layout: lane L -> row M=L&15; 16 halves cover K = {kb..kb+7, kb+16..kb+23},
//           kb = (L>=16)?8:0  -> two contiguous 8-half (16B) runs.
// B layout: lane L -> col N=L&15; 16 halves cover K = {kb..kb+7, kb+16..kb+23}, kb=(L>=16)?8:0
//           -> same K order as A: two contiguous 8-half (16B) runs.
// D layout: lane L -> col N=L&15; VGPR r -> row M = r + ((L>=16)?8:0).
// -------------------------------------------------------------------------
__device__ __forceinline__ v8f wmma_f16(v16h a, v16h b, v8f c) {
  return __builtin_amdgcn_wmma_f32_16x16x32_f16(false, a, false, b,
                                                (short)0, c, false, false);
}

// A fragment from a row-major [rows x pitch] f16 matrix (tile rows row0..row0+15,
// K-slice k0..k0+31).
__device__ __forceinline__ v16h load_a_frag(const half_t* p, int pitch,
                                            int row0, int k0, int lane) {
  const int m  = lane & 15;
  const int kb = (lane & 16) ? 8 : 0;
  const half_t* q = p + (size_t)(row0 + m) * pitch + k0 + kb;
  v8h lo = *(const v8h*)q;          // K = k0+kb .. +7
  v8h hi = *(const v8h*)(q + 16);   // K = k0+kb+16 .. +23
  v16h r;
#pragma unroll
  for (int i = 0; i < 8; ++i) { r[i] = lo[i]; r[i + 8] = hi[i]; }
  return r;
}

// B fragment where the logical B (K x N) is stored TRANSPOSED row-major as
// X[N][K] with row pitch `pitch`:  B(kk,N) = X[n0+N][k0+kk]  -> contiguous.
__device__ __forceinline__ v16h load_bt_frag(const half_t* p, int pitch,
                                             int n0, int k0, int lane) {
  const int nn = lane & 15;
  const int kb = (lane & 16) ? 8 : 0;
  const half_t* q = p + (size_t)(n0 + nn) * pitch + k0 + kb;
  v8h lo = *(const v8h*)q;
  v8h hi = *(const v8h*)(q + 16);
  v16h r;
#pragma unroll
  for (int i = 0; i < 8; ++i) { r[i] = lo[i]; r[i + 8] = hi[i]; }
  return r;
}

// =========================================================================
// Phase 1: per-chunk preprocessing. One block per (b,h,chunk).
//   - l2norm(q)*Dk^-1/2, l2norm(k)   (f32 -> f16)
//   - gg = within-chunk cumsum(g)
//   - S = Kn @ Kn^T (WMMA), attn = -(beta_i * S * exp(gg_i-gg_j)) strictly-lower
//   - UT forward substitution + I
//   - v_chunk = attn @ (v*beta)            (WMMA)
//   - k_cumdecay = attn @ (k*beta*e^gg)    (WMMA)
// Workspace out (per chunk): q,k (f16 row-major), k^T (f16), kcd (f16),
//                            v_chunk (f32), gg (f32)
// =========================================================================
__global__ void __launch_bounds__(256)
gdn_phase1(const float* __restrict__ Q, const float* __restrict__ Kin,
           const float* __restrict__ Vin, const float* __restrict__ G,
           const float* __restrict__ Beta,
           half_t* __restrict__ wq, half_t* __restrict__ wk,
           half_t* __restrict__ wkT, half_t* __restrict__ wkcd,
           float* __restrict__ wvc, float* __restrict__ wgg) {
  extern __shared__ char smem[];
  half_t* kf    = (half_t*)(smem);                 // [64][128]   16384 B
  float*  attnF = (float*) (smem + 16384);         // [64][64]    16384 B
  half_t* attnH = (half_t*)(smem + 32768);         // [64][64]     8192 B
  half_t* vbT   = (half_t*)(smem + 40960);         // [128][80]   20480 B
  half_t* kbgT  = (half_t*)(smem + 61440);         // [128][80]   20480 B
  float*  gg    = (float*) (smem + 81920);         // [64]
  float*  bv    = (float*) (smem + 82176);         // [64]

  const int blk = blockIdx.x;          // bh*NC + n
  const int n   = blk % NC;
  const int bh  = blk / NC;
  const int h   = bh % HH;
  const int b   = bh / HH;
  const int s0  = n * CC;
  const int tid = threadIdx.x;
  const int lane = tid & 31, wave = tid >> 5;
  const int nn   = lane & 15, mofs = (lane & 16) ? 8 : 0;

  // ---- load raw g/beta --------------------------------------------------
  if (tid < CC) {
    const size_t gi = (size_t)((b * SS + s0 + tid) * HH + h);
    gg[tid] = G[gi];
    bv[tid] = Beta[gi];
  }
  __syncthreads();

  // ---- cumsum (thread 0) + l2norm of q,k (4 threads per row) ------------
  if (tid == 0) {
    float acc = 0.f;
    for (int i = 0; i < CC; ++i) { acc += gg[i]; gg[i] = acc; }
  }
  {
    const int r = tid >> 2;           // row 0..63
    const int part = tid & 3;         // 32 elements each
    const size_t base =
        ((size_t)((b * SS + s0 + r) * HH + h)) * DK + part * 32;
    float qv[32], kv[32];
    float sq = 0.f, sk = 0.f;
#pragma unroll
    for (int j = 0; j < 32; ++j) {
      const float x = Q[base + j];  qv[j] = x;  sq += x * x;
      const float y = Kin[base + j]; kv[j] = y; sk += y * y;
    }
    sq += __shfl_xor(sq, 1, 32); sq += __shfl_xor(sq, 2, 32);
    sk += __shfl_xor(sk, 1, 32); sk += __shfl_xor(sk, 2, 32);
    const float qs = rsqrtf(sq + 1e-6f) * 0.08838834764831845f; // * Dk^-0.5
    const float ks = rsqrtf(sk + 1e-6f);
    half_t* wq_c  = wq  + (size_t)blk * CC * DK;
    half_t* wk_c  = wk  + (size_t)blk * CC * DK;
    half_t* wkT_c = wkT + (size_t)blk * CC * DK;
#pragma unroll
    for (int j = 0; j < 32; ++j) {
      const int d = part * 32 + j;
      const half_t qh = (half_t)(qv[j] * qs);
      const half_t kh = (half_t)(kv[j] * ks);
      wq_c[r * DK + d]  = qh;
      wk_c[r * DK + d]  = kh;
      wkT_c[d * CC + r] = kh;       // transposed copy for phase-2 state update
      kf[r * 128 + d]   = kh;
    }
  }
  __syncthreads();

  // ---- v*beta (transposed) and k*beta*exp(gg) (transposed), gg to ws ----
  {
    const int r = tid >> 2, part = tid & 3;
    const float bb = bv[r];
    const float eg = __expf(gg[r]);
    const size_t vbase =
        ((size_t)((b * SS + s0 + r) * HH + h)) * DV + part * 32;
#pragma unroll
    for (int j = 0; j < 32; ++j) {
      const int d = part * 32 + j;
      vbT[d * 80 + r]  = (half_t)(Vin[vbase + j] * bb);
      kbgT[d * 80 + r] = (half_t)((float)kf[r * 128 + d] * bb * eg);
    }
    if (tid < CC) wgg[(size_t)blk * CC + tid] = gg[tid];
  }
  __syncthreads();

  // ---- S = Kn @ Kn^T (16 tiles, 2 per wave), masked/decayed -> attnF ----
  for (int t = wave * 2; t < wave * 2 + 2; ++t) {
    const int ti = t >> 2, tj = t & 3;
    v8f acc = {};
#pragma unroll
    for (int k0 = 0; k0 < DK; k0 += 32) {
      v16h a = load_a_frag(kf, 128, ti * 16, k0, lane);
      v16h bb = load_bt_frag(kf, 128, tj * 16, k0, lane); // Kn^T contiguous
      acc = wmma_f16(a, bb, acc);
    }
#pragma unroll
    for (int r2 = 0; r2 < 8; ++r2) {
      const int i = ti * 16 + mofs + r2;
      const int j = tj * 16 + nn;
      float val = 0.f;
      if (i > j) val = -bv[i] * acc[r2] * __expf(gg[i] - gg[j]);
      attnF[i * 64 + j] = val;
    }
  }
  __syncthreads();

  // ---- UT transform (sequential forward substitution) -------------------
  for (int i = 1; i < CC; ++i) {
    float accu = 0.f;
    if (tid < CC) {
      accu = attnF[i * 64 + tid];
      for (int j = 0; j < i; ++j)
        accu += attnF[i * 64 + j] * attnF[j * 64 + tid];
    }
    __syncthreads();
    if (tid < CC) attnF[i * 64 + tid] = accu;
    __syncthreads();
  }
  if (tid < CC) attnF[tid * 64 + tid] = 1.0f;   // + I
  __syncthreads();
  for (int e = tid; e < 64 * 64; e += 256) attnH[e] = (half_t)attnF[e];
  __syncthreads();

  // ---- v_chunk = attn @ vb ; kcd = attn @ kbg  (32 tiles each, 4/wave) --
  {
    const int tm = wave >> 1, tn0 = (wave & 1) * 4;
    float*  wvc_c  = wvc  + (size_t)blk * CC * DV;
    half_t* wkcd_c = wkcd + (size_t)blk * CC * DK;
#pragma unroll
    for (int j = 0; j < 4; ++j) {
      const int tn = tn0 + j;
      v8f accv = {}, acck = {};
#pragma unroll
      for (int k0 = 0; k0 < CC; k0 += 32) {
        v16h a  = load_a_frag(attnH, 64, tm * 16, k0, lane);
        v16h b1 = load_bt_frag(vbT, 80, tn * 16, k0, lane);
        v16h b2 = load_bt_frag(kbgT, 80, tn * 16, k0, lane);
        accv = wmma_f16(a, b1, accv);
        acck = wmma_f16(a, b2, acck);
      }
#pragma unroll
      for (int r2 = 0; r2 < 8; ++r2) {
        const int i = tm * 16 + mofs + r2;
        const int d = tn * 16 + nn;
        wvc_c[i * DV + d]  = accv[r2];
        wkcd_c[i * DK + d] = (half_t)acck[r2];
      }
    }
  }
}

// =========================================================================
// Phase 2: sequential scan over 64 chunks per (b,h). One block per (b,h).
// State (128x128 f32) lives in WMMA accumulators: wave w owns rows w*16..+15.
// stateT (f16, [v][k]) in LDS is the B-operand for the @state matmuls.
// =========================================================================
__global__ void __launch_bounds__(256)
gdn_phase2(const half_t* __restrict__ wq, const half_t* __restrict__ wk,
           const half_t* __restrict__ wkT, const half_t* __restrict__ wkcd,
           const float* __restrict__ wvc, const float* __restrict__ wgg,
           float* __restrict__ Out) {
  extern __shared__ char smem[];
  half_t* stateT = (half_t*)(smem);                  // [128][144]  36864 B
  half_t* vnT    = (half_t*)(smem + 36864);          // [128][80]   20480 B
  half_t* qkH    = (half_t*)(smem + 57344);          // [64][64]     8192 B
  float*  gg     = (float*) (smem + 65536);          // [64]
  float*  egi    = (float*) (smem + 65792);          // exp(gg_i)
  float*  ewi    = (float*) (smem + 66048);          // exp(g_last-gg_i)

  const int bh = blockIdx.x;
  const int h = bh % HH, b = bh / HH;
  const int tid = threadIdx.x;
  const int lane = tid & 31, wave = tid >> 5;
  const int nn = lane & 15, mofs = (lane & 16) ? 8 : 0;

  for (int i = tid; i < 128 * 144; i += 256) stateT[i] = (half_t)0.f;

  v8f st[8];                                         // state tiles (row-stripe)
#pragma unroll
  for (int t = 0; t < 8; ++t)
#pragma unroll
    for (int r2 = 0; r2 < 8; ++r2) st[t][r2] = 0.f;
  __syncthreads();

  for (int n = 0; n < NC; ++n) {
    const size_t blk = (size_t)bh * NC + n;
    const half_t* q_c   = wq   + blk * CC * DK;
    const half_t* k_c   = wk   + blk * CC * DK;
    const half_t* kT_c  = wkT  + blk * CC * DK;
    const half_t* kcd_c = wkcd + blk * CC * DK;
    const float*  vc_c  = wvc  + blk * CC * DV;
    const float*  gg_c  = wgg  + blk * CC;

    if (n + 1 < NC) {                                 // global_prefetch_b8
      __builtin_prefetch(wq  + (blk + 1) * CC * DK, 0, 0);
      __builtin_prefetch(wk  + (blk + 1) * CC * DK, 0, 0);
      __builtin_prefetch(wkcd + (blk + 1) * CC * DK, 0, 0);
      __builtin_prefetch(wvc + (blk + 1) * CC * DV, 0, 0);
    }

    if (tid < CC) gg[tid] = gg_c[tid];
    __syncthreads();
    if (tid < CC) {
      egi[tid] = __expf(gg[tid]);
      ewi[tid] = __expf(gg[CC - 1] - gg[tid]);
    }
    __syncthreads();

    // ---- qk = (q @ k^T) * decay(incl diag)  -> qkH (f16)  [2 tiles/wave]
    for (int t = wave * 2; t < wave * 2 + 2; ++t) {
      const int ti = t >> 2, tj = t & 3;
      v8f acc = {};
#pragma unroll
      for (int k0 = 0; k0 < DK; k0 += 32) {
        v16h a = load_a_frag(q_c, 128, ti * 16, k0, lane);
        v16h bb = load_bt_frag(k_c, 128, tj * 16, k0, lane);
        acc = wmma_f16(a, bb, acc);
      }
#pragma unroll
      for (int r2 = 0; r2 < 8; ++r2) {
        const int i = ti * 16 + mofs + r2;
        const int j = tj * 16 + nn;
        const float v = (i >= j) ? acc[r2] * __expf(gg[i] - gg[j]) : 0.f;
        qkH[i * 64 + j] = (half_t)v;
      }
    }

    // ---- v_new = v_chunk - kcd @ state  -> vnT (f16, transposed) [4/wave]
    {
      const int tm = wave >> 1, tn0 = (wave & 1) * 4;
#pragma unroll
      for (int j = 0; j < 4; ++j) {
        const int tn = tn0 + j;
        v8f acc = {};
#pragma unroll
        for (int k0 = 0; k0 < DK; k0 += 32) {
          v16h a = load_a_frag(kcd_c, 128, tm * 16, k0, lane);
          v16h bb = load_bt_frag(stateT, 144, tn * 16, k0, lane);
          acc = wmma_f16(a, bb, acc);
        }
#pragma unroll
        for (int r2 = 0; r2 < 8; ++r2) {
          const int i = tm * 16 + mofs + r2;
          const int d = tn * 16 + nn;
          vnT[d * 80 + i] = (half_t)(vc_c[i * DV + d] - acc[r2]);
        }
      }
    }
    __syncthreads();

    // ---- out = (q*e^gg) @ state + qk @ v_new  -> global [4 tiles/wave] --
    {
      const int tm = wave >> 1, tn0 = (wave & 1) * 4;
      const size_t obase = ((size_t)((b * SS + n * CC) * HH + h)) * DV;
#pragma unroll
      for (int j = 0; j < 4; ++j) {
        const int tn = tn0 + j;
        v8f acc = {};
        const half_t sc = (half_t)egi[tm * 16 + nn];  // per-lane row scale
#pragma unroll
        for (int k0 = 0; k0 < DK; k0 += 32) {
          v16h a = load_a_frag(q_c, 128, tm * 16, k0, lane);
          a = a * sc;
          v16h bb = load_bt_frag(stateT, 144, tn * 16, k0, lane);
          acc = wmma_f16(a, bb, acc);
        }
#pragma unroll
        for (int k0 = 0; k0 < CC; k0 += 32) {
          v16h a = load_a_frag(qkH, 64, tm * 16, k0, lane);
          v16h bb = load_bt_frag(vnT, 80, tn * 16, k0, lane);
          acc = wmma_f16(a, bb, acc);
        }
#pragma unroll
        for (int r2 = 0; r2 < 8; ++r2) {
          const int i = tm * 16 + mofs + r2;
          const int d = tn * 16 + nn;
          Out[obase + (size_t)i * (HH * DV) + d] = acc[r2];
        }
      }
    }

    // ---- state = state*e^g_last + k_w^T @ v_new (accumulators) ----------
    {
      const float elast = __expf(gg[CC - 1]);
#pragma unroll
      for (int t = 0; t < 8; ++t)
#pragma unroll
        for (int r2 = 0; r2 < 8; ++r2) st[t][r2] *= elast;
      const int kb = (lane & 16) ? 8 : 0;
#pragma unroll
      for (int k0 = 0; k0 < CC; k0 += 32) {
        v16h a = load_a_frag(kT_c, 64, wave * 16, k0, lane); // k^T rows = d
#pragma unroll
        for (int hh = 0; hh < 16; ++hh) {           // per-half decay weight
          const int kk = k0 + kb + (hh < 8 ? hh : hh + 8);
          a[hh] = a[hh] * (half_t)ewi[kk];
        }
#pragma unroll
        for (int t = 0; t < 8; ++t) {
          v16h bb = load_bt_frag(vnT, 80, t * 16, k0, lane);
          st[t] = wmma_f16(a, bb, st[t]);
        }
      }
    }
    __syncthreads();                     // everyone done reading old stateT
#pragma unroll
    for (int t = 0; t < 8; ++t)
#pragma unroll
      for (int r2 = 0; r2 < 8; ++r2) {
        const int kd = wave * 16 + mofs + r2;        // state row (k)
        const int vd = t * 16 + nn;                  // state col (v)
        stateT[vd * 144 + kd] = (half_t)st[t][r2];
      }
    __syncthreads();                     // new stateT visible for next chunk
  }
}

// =========================================================================
extern "C" void kernel_launch(void* const* d_in, const int* in_sizes, int n_in,
                              void* d_out, int out_size, void* d_ws,
                              size_t ws_size, hipStream_t stream) {
  const float* Q    = (const float*)d_in[0];
  const float* Kin  = (const float*)d_in[1];
  const float* Vin  = (const float*)d_in[2];
  const float* G    = (const float*)d_in[3];
  const float* Beta = (const float*)d_in[4];
  // d_in[5] = chunk_size (fixed at 64 for this problem)
  float* Out = (float*)d_out;

  const size_t nchunks = (size_t)NBH * NC;           // 2048
  char* ws = (char*)d_ws;
  half_t* wq   = (half_t*)ws;  ws += nchunks * CC * DK * sizeof(half_t);
  half_t* wk   = (half_t*)ws;  ws += nchunks * CC * DK * sizeof(half_t);
  half_t* wkT  = (half_t*)ws;  ws += nchunks * CC * DK * sizeof(half_t);
  half_t* wkcd = (half_t*)ws;  ws += nchunks * CC * DK * sizeof(half_t);
  float*  wvc  = (float*)ws;   ws += nchunks * CC * DV * sizeof(float);
  float*  wgg  = (float*)ws;

  const size_t smem1 = 82432;   // phase-1 dynamic LDS
  const size_t smem2 = 66304;   // phase-2 dynamic LDS

  gdn_phase1<<<dim3(NBH * NC), dim3(256), smem1, stream>>>(
      Q, Kin, Vin, G, Beta, wq, wk, wkT, wkcd, wvc, wgg);
  gdn_phase2<<<dim3(NBH), dim3(256), smem2, stream>>>(
      wq, wk, wkT, wkcd, wvc, wgg, Out);
}
